// PatchDiscriminator_20469814133478
// MI455X (gfx1250) — matrix operand order repaired and checked by reading, not compiled, not run on hardware
//
#include <hip/hip_runtime.h>
#include <hip/hip_bf16.h>

typedef _Float16 f16;
typedef __attribute__((ext_vector_type(8)))  _Float16 v8h;   // 16 bytes
typedef __attribute__((ext_vector_type(16))) _Float16 v16h;  // 32 bytes (WMMA A/B frag)
typedef __attribute__((ext_vector_type(8)))  float    v8f;   // WMMA C/D frag

#define S_LEN 2048
#define HID   128
#define KDIM  384          // 3 * 128 im2col contraction
#define NK    682          // S_LEN / 3 chunks
#define EPSV  1e-5f

union Frag { v16h v; v8h h[2]; };

// ---------------------------------------------------------------------------
// Kernel 1: convert (O, I, T) f32 weights -> (O, k = t*128 + ic) f16
// so A-fragments are contiguous 16B runs matching the WMMA 16-bit A layout.
// ---------------------------------------------------------------------------
__global__ void wconvert_kernel(const float* __restrict__ w1,
                                const float* __restrict__ w2,
                                const float* __restrict__ wp,
                                f16* __restrict__ w1f,
                                f16* __restrict__ w2f,
                                f16* __restrict__ wpf) {
    int idx = blockIdx.x * blockDim.x + threadIdx.x;
    const int per = HID * KDIM;          // 49152
    if (idx >= 3 * per) return;
    int mat = idx / per;
    int r   = idx % per;
    int o   = r / KDIM;
    int k   = r % KDIM;
    int t   = k / HID;
    int ic  = k % HID;
    const float* src = (mat == 0) ? w1 : (mat == 1 ? w2 : wp);
    f16*         dst = (mat == 0) ? w1f : (mat == 1 ? w2f : wpf);
    dst[r] = (f16)src[o * KDIM + ic * 3 + t];
}

// ---------------------------------------------------------------------------
// Kernel 2: conv0 (1 -> 128 channels) + ReLU + BN0, output f16 [b][s][128]
// ---------------------------------------------------------------------------
__global__ void conv0_bn_kernel(const float* __restrict__ x,
                                const float* __restrict__ w0,   // (128,1,3)
                                const float* __restrict__ b0,
                                const float* __restrict__ gamma,
                                const float* __restrict__ beta,
                                const float* __restrict__ mean,
                                const float* __restrict__ var,
                                f16* __restrict__ h0) {
    int tid = threadIdx.x;
    int b   = blockIdx.y;
    int pos = blockIdx.x * 2 + (tid >> 7);
    int ch  = tid & (HID - 1);

    const float* xb = x + (size_t)b * S_LEN;
    float xm1 = (pos > 0)         ? xb[pos - 1] : 0.f;
    float x00 = xb[pos];
    float xp1 = (pos < S_LEN - 1) ? xb[pos + 1] : 0.f;

    float y = w0[ch * 3 + 0] * xm1 + w0[ch * 3 + 1] * x00 + w0[ch * 3 + 2] * xp1 + b0[ch];
    y = fmaxf(y, 0.f);
    float sc = gamma[ch] * __frsqrt_rn(var[ch] + EPSV);
    y = (y - mean[ch]) * sc + beta[ch];
    h0[((size_t)b * S_LEN + pos) * HID + ch] = (f16)y;
}

// ---------------------------------------------------------------------------
// Kernel 3: 128->128 conv1d(k=3) as WMMA GEMM, fused bias+ReLU+BN epilogue.
// Block = 256 threads (8 waves). Wave w owns output channels [16w, 16w+16).
// Block covers 128 output positions; activations (with pad halo) staged in LDS.
// Contraction: 384 = 12 steps of v_wmma_f32_16x16x32_f16.
// ---------------------------------------------------------------------------
__global__ void conv_gemm_kernel(const f16* __restrict__ hin,   // [b][s][128] f16
                                 const f16* __restrict__ wf,    // [o][k] f16 (k = t*128+ic)
                                 const float* __restrict__ bias,
                                 const float* __restrict__ gamma,
                                 const float* __restrict__ beta,
                                 const float* __restrict__ mean,
                                 const float* __restrict__ var,
                                 f16* __restrict__ hout) {
    __shared__ f16 act[130 * HID];       // positions p0-1 .. p0+128, 33.3 KB

    const int tid = threadIdx.x;
    const int b   = blockIdx.y;
    const int p0  = blockIdx.x * 128;

    // ---- stage activation tile (+halo) into LDS, 16B vectors ----
    for (int g = tid; g < 130 * 16; g += 256) {
        int row = g >> 4;
        int c8  = (g & 15) << 3;
        int pos = p0 - 1 + row;
        v8h v = {};
        if (pos >= 0 && pos < S_LEN)
            v = *(const v8h*)(hin + ((size_t)b * S_LEN + pos) * HID + c8);
        *(v8h*)(act + row * HID + c8) = v;
    }
    __syncthreads();

    const int wave   = tid >> 5;
    const int lane   = tid & 31;
    const int mrow   = lane & 15;        // A row / B column within tile
    const int half   = lane >> 4;        // K half selector
    const int ocbase = wave * 16;

    // per-lane epilogue constants (oc = ocbase + r + half*8)
    float scv[8], biv[8], mnv[8], btv[8];
#pragma unroll
    for (int r = 0; r < 8; ++r) {
        int oc = ocbase + r + half * 8;
        scv[r] = gamma[oc] * __frsqrt_rn(var[oc] + EPSV);
        biv[r] = bias[oc];
        mnv[r] = mean[oc];
        btv[r] = beta[oc];
    }

    for (int pt = 0; pt < 8; ++pt) {
        v8f acc = {};
#pragma unroll
        for (int ks = 0; ks < 12; ++ks) {
            Frag A, B;
            // A: weights, rows = out channels, K offset half*8 (ISA 16-bit A layout)
            const f16* ap = wf + (ocbase + mrow) * KDIM + ks * 32 + half * 8;
            A.h[0] = *(const v8h*)ap;
            A.h[1] = *(const v8h*)(ap + 16);
            // B: im2col activations, K offset half*8 (ISA 16-bit B layout)
            int kb  = ks * 32 + half * 8;
            int tap = kb >> 7;
            int ic0 = kb & (HID - 1);
            int row = pt * 16 + mrow + tap;          // halo-shifted input row
            const f16* bp = act + row * HID + ic0;
            B.h[0] = *(const v8h*)bp;
            B.h[1] = *(const v8h*)(bp + 16);
            acc = __builtin_amdgcn_wmma_f32_16x16x32_f16(
                false, A.v, false, B.v, (short)0, acc, false, false);
        }
        // epilogue: bias + ReLU + BN, store f16
        size_t pos = (size_t)p0 + pt * 16 + mrow;
        f16* op = hout + ((size_t)b * S_LEN + pos) * HID;
#pragma unroll
        for (int r = 0; r < 8; ++r) {
            int oc  = ocbase + r + half * 8;
            float y = fmaxf(acc[r] + biv[r], 0.f);
            y = (y - mnv[r]) * scv[r] + btv[r];
            op[oc] = (f16)y;
        }
    }
}

// ---------------------------------------------------------------------------
// Kernel 4: chunked projection z[b,k,o] = relu(sum_{t,h} h2[b,3k+t,h]*W[o,h,t] + b)
// Same 128 x 384 x N WMMA GEMM; B-fragments straight from global (stride-3 rows).
// ---------------------------------------------------------------------------
__global__ void pre_gemm_kernel(const f16* __restrict__ h2,    // [b][s][128]
                                const f16* __restrict__ wpf,   // [o][k] f16
                                const float* __restrict__ pre_b,
                                f16* __restrict__ z) {         // [b][682][128]
    const int tid = threadIdx.x;
    const int b   = blockIdx.y;
    const int k0  = blockIdx.x * 128;

    const int wave   = tid >> 5;
    const int lane   = tid & 31;
    const int mrow   = lane & 15;
    const int half   = lane >> 4;
    const int ocbase = wave * 16;

    for (int pt = 0; pt < 8; ++pt) {
        int kk  = k0 + pt * 16 + mrow;               // chunk index (column)
        int kkc = (kk < NK) ? kk : (NK - 1);         // clamp for loads
        v8f acc = {};
#pragma unroll
        for (int ks = 0; ks < 12; ++ks) {
            Frag A, B;
            const f16* ap = wpf + (ocbase + mrow) * KDIM + ks * 32 + half * 8;
            A.h[0] = *(const v8h*)ap;
            A.h[1] = *(const v8h*)(ap + 16);
            int kb  = ks * 32 + half * 8;
            int tap = kb >> 7;
            int ic0 = kb & (HID - 1);
            int pos = 3 * kkc + tap;                 // <= 2045 < 2048
            const f16* bp = h2 + ((size_t)b * S_LEN + pos) * HID + ic0;
            B.h[0] = *(const v8h*)bp;
            B.h[1] = *(const v8h*)(bp + 16);
            acc = __builtin_amdgcn_wmma_f32_16x16x32_f16(
                false, A.v, false, B.v, (short)0, acc, false, false);
        }
        if (kk < NK) {
            f16* op = z + ((size_t)b * NK + kk) * HID;
#pragma unroll
            for (int r = 0; r < 8; ++r) {
                int oc  = ocbase + r + half * 8;
                float y = fmaxf(acc[r] + pre_b[oc], 0.f);
                op[oc] = (f16)y;
            }
        }
    }
}

// ---------------------------------------------------------------------------
// Kernel 5: final 128 -> 1 dot + seq_len masking, f32 output
// ---------------------------------------------------------------------------
__global__ void fc_mask_kernel(const f16* __restrict__ z,
                               const float* __restrict__ fc_w,
                               const float* __restrict__ fc_b,
                               const long long* __restrict__ seq_lens,
                               float* __restrict__ out) {
    int idx = blockIdx.x * blockDim.x + threadIdx.x;
    if (idx >= 64 * NK) return;
    int b  = idx / NK;
    int kk = idx % NK;
    long long nch = seq_lens[b] / 3;

    const f16* zp = z + (size_t)idx * HID;
    float s = 0.f;
#pragma unroll
    for (int c = 0; c < HID; c += 8) {
        v8h v = *(const v8h*)(zp + c);
#pragma unroll
        for (int j = 0; j < 8; ++j) s += (float)v[j] * fc_w[c + j];
    }
    out[idx] = (kk < (int)nch) ? (s + fc_b[0]) : 0.f;
}

// ---------------------------------------------------------------------------
extern "C" void kernel_launch(void* const* d_in, const int* in_sizes, int n_in,
                              void* d_out, int out_size, void* d_ws, size_t ws_size,
                              hipStream_t stream) {
    const float*     x        = (const float*)d_in[0];
    const long long* seq_lens = (const long long*)d_in[1];   // int64
    const float*     conv_w0  = (const float*)d_in[2];
    const float*     conv_b0  = (const float*)d_in[3];
    const float*     conv_w1  = (const float*)d_in[4];
    const float*     conv_b1  = (const float*)d_in[5];
    const float*     conv_w2  = (const float*)d_in[6];
    const float*     conv_b2  = (const float*)d_in[7];
    const float*     bn_gamma = (const float*)d_in[8];       // (3,128)
    const float*     bn_beta  = (const float*)d_in[9];
    const float*     bn_mean  = (const float*)d_in[10];
    const float*     bn_var   = (const float*)d_in[11];
    const float*     pre_w    = (const float*)d_in[12];
    const float*     pre_b    = (const float*)d_in[13];
    const float*     fc_w     = (const float*)d_in[14];
    const float*     fc_b     = (const float*)d_in[15];
    float*           out      = (float*)d_out;

    char* ws = (char*)d_ws;
    const size_t ACT_BYTES = (size_t)64 * S_LEN * HID * sizeof(f16); // 33,554,432
    f16* h0  = (f16*)(ws);
    f16* h1  = (f16*)(ws + ACT_BYTES);
    f16* h2  = (f16*)(ws + 2 * ACT_BYTES);
    f16* zz  = (f16*)(ws + 3 * ACT_BYTES);                    // 64*682*128 f16
    char* wb = ws + 3 * ACT_BYTES + (size_t)64 * NK * HID * sizeof(f16);
    f16* w1f = (f16*)(wb);
    f16* w2f = (f16*)(wb + 98304);
    f16* wpf = (f16*)(wb + 2 * 98304);

    // 1) weight repack to f16 im2col-K layout
    wconvert_kernel<<<dim3((3 * HID * KDIM + 255) / 256), dim3(256), 0, stream>>>(
        conv_w1, conv_w2, pre_w, w1f, w2f, wpf);

    // 2) conv0 + ReLU + BN0 (elementwise per channel)
    conv0_bn_kernel<<<dim3(S_LEN / 2, 64), dim3(256), 0, stream>>>(
        x, conv_w0, conv_b0, bn_gamma + 0, bn_beta + 0, bn_mean + 0, bn_var + 0, h0);

    // 3) conv1 as WMMA GEMM + fused ReLU/BN1
    conv_gemm_kernel<<<dim3(S_LEN / 128, 64), dim3(256), 0, stream>>>(
        h0, w1f, conv_b1, bn_gamma + HID, bn_beta + HID, bn_mean + HID, bn_var + HID, h1);

    // 4) conv2 as WMMA GEMM + fused ReLU/BN2
    conv_gemm_kernel<<<dim3(S_LEN / 128, 64), dim3(256), 0, stream>>>(
        h1, w2f, conv_b2, bn_gamma + 2 * HID, bn_beta + 2 * HID,
        bn_mean + 2 * HID, bn_var + 2 * HID, h2);

    // 5) chunked projection as WMMA GEMM + fused ReLU
    pre_gemm_kernel<<<dim3((NK + 127) / 128, 64), dim3(256), 0, stream>>>(
        h2, wpf, pre_b, zz);

    // 6) final dot + masking
    fc_mask_kernel<<<dim3((64 * NK + 255) / 256), dim3(256), 0, stream>>>(
        zz, fc_w, fc_b, seq_lens, out);
}
